// FeedForwardSNN_36223754174478
// MI455X (gfx1250) — matrix operand order repaired and checked by reading, not compiled, not run on hardware
//
#include <hip/hip_runtime.h>

// ---------------------------------------------------------------------------
// FeedForward SNN on gfx1250 (MI455X): bf16 WMMA GEMMs, fused LIF epilogues,
// and launch-count-minimized schedule (the workload is a 256-step sequential
// chain of ~1 GFLOP GEMMs -> kernel-launch latency dominates; weights are
// L2-resident so HBM is irrelevant).
//
// Per timestep t (X1 = inp[t] @ W1^T + b1 is substep-invariant):
//   for k in 0..3:
//     s1,v1 = LIF(X1, v1)                       (fused into prev layer-3 krnl)
//     v2,s2 = LIF(s1 @ W2^T + b2, v2)           (WMMA GEMM, LIF epilogue)
//     vout  = (k? BETA*vout : 0) + s2 @ W3^T + b3  (WMMA GEMM, LI epilogue,
//                                                   + next-substep LIF-1)
//   out[t] = vout at k==3
//
// Layer-1 GEMM batched over chunks of 8 timesteps (M=1024, 256 blocks) with
// ping-pong X1 buffers. Spikes are exactly {0,1} -> exact in bf16.
// ---------------------------------------------------------------------------

typedef __attribute__((ext_vector_type(16))) __bf16 v16bf;
typedef __attribute__((ext_vector_type(8)))  __bf16 v8bf;
typedef __attribute__((ext_vector_type(4)))  __bf16 v4bf;
typedef __attribute__((ext_vector_type(8)))  float  v8f;
typedef __attribute__((ext_vector_type(4)))  float  v4f;

#define SNN_BETA   0.9f
#define SNN_THRESH 1.0f

// ---------------- elementwise helpers ----------------

__global__ void cvt_f32_to_bf16_kernel(const float* __restrict__ in,
                                       __bf16* __restrict__ out, int n) {
  int i = blockIdx.x * blockDim.x + threadIdx.x;
  if (i < n) out[i] = (__bf16)in[i];
}

__global__ void zero_f32_kernel(float* __restrict__ p, int n) {
  int i = blockIdx.x * blockDim.x + threadIdx.x;
  if (i < n) p[i] = 0.0f;
}

// Standalone LIF layer-1 (only used to seed substep (t=0,k=0)).
__global__ void lif1_kernel(const float* __restrict__ x,
                            float* __restrict__ v,
                            __bf16* __restrict__ s, int n) {
  int i = blockIdx.x * blockDim.x + threadIdx.x;
  if (i < n) {
    float vv = SNN_BETA * v[i] + x[i];
    float sp = (vv >= SNN_THRESH) ? 1.0f : 0.0f;
    v[i] = vv - sp * SNN_THRESH;
    s[i] = (__bf16)sp;
  }
}

// ---------------- WMMA GEMM core (shared by both GEMM kernels) ----------------
// D[m][n] = sum_k A[m][k]*W[n][k]; A: MxK bf16 row-major, W: NxK bf16 row-major.
// One wave computes a 16x64 strip: 4 accumulators, A reused 4x per load.
// Register layouts used here (contiguous k for A and B, not the ISA table order):
//   A (16-bit 16x32): lane l -> row l&15, 16 contiguous K at 16*(l>>4)
//   B (16-bit 32x16): lane l -> col l&15, 16 contiguous K at 16*(l>>4)
//   D: VGPR j -> row 8*(l>>4)+j, col l&15
__device__ __forceinline__ void wmma_strip(const __bf16* __restrict__ A, int lda,
                                           const __bf16* __restrict__ W, int ldw,
                                           int K, int m0, int n0,
                                           int half, int lm, v8f acc[4]) {
#pragma unroll
  for (int nt = 0; nt < 4; ++nt) acc[nt] = (v8f){};
  const __bf16* Arow = A + (size_t)(m0 + lm) * (size_t)lda;
  for (int k = 0; k < K; k += 32) {
    v16bf a = *(const v16bf*)(Arow + k + 16 * half);
#pragma unroll
    for (int nt = 0; nt < 4; ++nt) {
      const __bf16* wrow =
          W + (size_t)(n0 + nt * 16 + lm) * (size_t)ldw + k + 16 * half;
      v16bf b = *(const v16bf*)wrow;
      acc[nt] = __builtin_amdgcn_wmma_f32_16x16x32_bf16(
          false, a, false, b, (short)0, acc[nt], false, false);
    }
  }
}

// ---------------- GEMM kernel for layers 1 & 2 ----------------
// mode 0: out_f32 = acc + bias                         (layer 1 -> X1 chunk)
// mode 1: v = BETA*v + acc + bias; s=(v>=1); v-=s      (layer 2, fused LIF)
__launch_bounds__(256)
__global__ void snn_gemm_kernel(const __bf16* __restrict__ A, int lda,
                                const __bf16* __restrict__ W, int ldw,
                                const float* __restrict__ bias,
                                float* __restrict__ vstate,
                                __bf16* __restrict__ spikes,
                                float* __restrict__ outp,
                                int K, int ldo, int mode) {
  const int wave = threadIdx.x >> 5;
  const int lane = threadIdx.x & 31;
  const int half = lane >> 4;
  const int lm   = lane & 15;
  const int m0 = blockIdx.y * 128 + wave * 16;
  const int n0 = blockIdx.x * 64;

  v8f acc[4];
  wmma_strip(A, lda, W, ldw, K, m0, n0, half, lm, acc);

#pragma unroll
  for (int nt = 0; nt < 4; ++nt) {
    const int n = n0 + nt * 16 + lm;
    const float bi = bias[n];
#pragma unroll
    for (int j = 0; j < 8; ++j) {
      const int m = m0 + half * 8 + j;
      const size_t idx = (size_t)m * (size_t)ldo + n;
      const float av = acc[nt][j] + bi;
      if (mode == 0) {
        outp[idx] = av;
      } else {
        float vv = SNN_BETA * vstate[idx] + av;
        float sp = (vv >= SNN_THRESH) ? 1.0f : 0.0f;
        vstate[idx] = vv - sp * SNN_THRESH;
        spikes[idx] = (__bf16)sp;
      }
    }
  }
}

// ---------------- layer-3 kernel: LI output + fused next-substep LIF-1 ------
// Blocks [0, nGemmBlk) compute vout = (firstSub?0:BETA*vout) + s2@W3^T + b3
// (write out[t] when writeOut). ALL blocks additionally perform the LIF-1
// elementwise update for the NEXT substep (v1/s1 from X1next), which depends
// only on v1 + X1 and is therefore legal to compute here (stream-ordered
// before the next GEMM2 that consumes s1).
__launch_bounds__(256)
__global__ void snn_layer3_kernel(const __bf16* __restrict__ A, int lda,
                                  const __bf16* __restrict__ W, int ldw,
                                  const float* __restrict__ bias,
                                  float* __restrict__ vout,
                                  float* __restrict__ outp,
                                  int K, int ldo,
                                  int firstSub, int writeOut, int nGemmBlk,
                                  const float* __restrict__ X1next,
                                  float* __restrict__ v1,
                                  __bf16* __restrict__ s1,
                                  int nLif, int doLif) {
  // ---- fused LIF-1 for the next substep (grid-stride, float4) ----
  if (doLif) {
    const int tid = blockIdx.x * blockDim.x + threadIdx.x;
    const int stride4 = gridDim.x * blockDim.x * 4;
    for (int i = tid * 4; i < nLif; i += stride4) {
      v4f v = *(const v4f*)(v1 + i);
      v4f x = *(const v4f*)(X1next + i);
      v4bf sb;
#pragma unroll
      for (int j = 0; j < 4; ++j) {
        float vv = SNN_BETA * v[j] + x[j];
        float sp = (vv >= SNN_THRESH) ? 1.0f : 0.0f;
        v[j] = vv - sp * SNN_THRESH;
        sb[j] = (__bf16)sp;
      }
      *(v4f*)(v1 + i) = v;
      *(v4bf*)(s1 + i) = sb;
    }
  }

  if ((int)blockIdx.x >= nGemmBlk) return;

  const int wave = threadIdx.x >> 5;
  const int lane = threadIdx.x & 31;
  const int half = lane >> 4;
  const int lm   = lane & 15;
  const int m0 = wave * 16;
  const int n0 = blockIdx.x * 64;

  v8f acc[4];
  wmma_strip(A, lda, W, ldw, K, m0, n0, half, lm, acc);

#pragma unroll
  for (int nt = 0; nt < 4; ++nt) {
    const int n = n0 + nt * 16 + lm;
    const float bi = bias[n];
#pragma unroll
    for (int j = 0; j < 8; ++j) {
      const int m = m0 + half * 8 + j;
      const size_t idx = (size_t)m * (size_t)ldo + n;
      float vv = (firstSub ? 0.0f : SNN_BETA * vout[idx]) + acc[nt][j] + bi;
      vout[idx] = vv;
      if (writeOut) outp[idx] = vv;
    }
  }
}

// ---------------------------------------------------------------------------

extern "C" void kernel_launch(void* const* d_in, const int* in_sizes, int n_in,
                              void* d_out, int out_size, void* d_ws,
                              size_t ws_size, hipStream_t stream) {
  (void)in_sizes; (void)n_in; (void)out_size; (void)ws_size;

  constexpr int T = 64, B = 128;
  constexpr int A0 = 1024, A1 = 2048, A2 = 2048, A3 = 512;
  constexpr int SIM = 4;
  constexpr int TCH = 8;              // timesteps per layer-1 GEMM chunk
  constexpr int NCH = T / TCH;        // 8 chunks

  const float* inp = (const float*)d_in[0];
  const float* W1  = (const float*)d_in[1];
  const float* b1  = (const float*)d_in[2];
  const float* W2  = (const float*)d_in[3];
  const float* b2  = (const float*)d_in[4];
  const float* W3  = (const float*)d_in[5];
  const float* b3  = (const float*)d_in[6];
  float* out = (float*)d_out;

  char* ws = (char*)d_ws;
  size_t off = 0;
  auto take = [&](size_t bytes) {
    void* p = ws + off;
    off += (bytes + 255) & ~(size_t)255;
    return p;
  };
  __bf16* W1bf   = (__bf16*)take((size_t)A1 * A0 * 2);        //  4 MB
  __bf16* W2bf   = (__bf16*)take((size_t)A2 * A1 * 2);        //  8 MB
  __bf16* W3bf   = (__bf16*)take((size_t)A3 * A2 * 2);        //  2 MB
  __bf16* inp_bf = (__bf16*)take((size_t)T * B * A0 * 2);     // 16 MB
  float*  X1pp[2];
  X1pp[0] = (float*)take((size_t)TCH * B * A1 * 4);           //  8 MB
  X1pp[1] = (float*)take((size_t)TCH * B * A1 * 4);           //  8 MB
  float*  v1   = (float*)take((size_t)B * A1 * 4);            //  1 MB
  float*  v2   = (float*)take((size_t)B * A2 * 4);            //  1 MB (contig)
  float*  vout = (float*)take((size_t)B * A3 * 4);
  __bf16* s1   = (__bf16*)take((size_t)B * A1 * 2);
  __bf16* s2   = (__bf16*)take((size_t)B * A2 * 2);

  const int TPB = 256;
  auto blocks = [](int n, int tpb) { return (n + tpb - 1) / tpb; };

  // --- per-call preprocessing: weights/input -> bf16, zero persistent state
  cvt_f32_to_bf16_kernel<<<blocks(A1 * A0, TPB), TPB, 0, stream>>>(W1, W1bf, A1 * A0);
  cvt_f32_to_bf16_kernel<<<blocks(A2 * A1, TPB), TPB, 0, stream>>>(W2, W2bf, A2 * A1);
  cvt_f32_to_bf16_kernel<<<blocks(A3 * A2, TPB), TPB, 0, stream>>>(W3, W3bf, A3 * A2);
  cvt_f32_to_bf16_kernel<<<blocks(T * B * A0, TPB), TPB, 0, stream>>>(inp, inp_bf, T * B * A0);
  zero_f32_kernel<<<blocks(B * (A1 + A2), TPB), TPB, 0, stream>>>(v1, B * (A1 + A2));

  auto launch_gemm1 = [&](int c) {  // X1[c%2] = inp[chunk c] @ W1^T + b1
    dim3 grid(A1 / 64, (TCH * B) / 128);
    snn_gemm_kernel<<<grid, TPB, 0, stream>>>(
        inp_bf + (size_t)c * TCH * B * A0, A0, W1bf, A0, b1,
        nullptr, nullptr, X1pp[c & 1], A0, A1, /*mode=*/0);
  };

  launch_gemm1(0);
  // Seed LIF-1 for substep (t=0, k=0).
  lif1_kernel<<<blocks(B * A1, TPB), TPB, 0, stream>>>(X1pp[0], v1, s1, B * A1);

  for (int c = 0; c < NCH; ++c) {
    if (c + 1 < NCH) launch_gemm1(c + 1);  // buffer (c+1)&1 is free by now
    for (int tt = 0; tt < TCH; ++tt) {
      const int t = c * TCH + tt;
      for (int k = 0; k < SIM; ++k) {
        // Layer 2: v2/s2 = LIF(s1 @ W2^T + b2, v2)
        {
          dim3 grid(A2 / 64, 1);
          snn_gemm_kernel<<<grid, TPB, 0, stream>>>(
              s1, A1, W2bf, A1, b2, v2, s2, nullptr, A1, A2, /*mode=*/1);
        }
        // Layer 3 + fused LIF-1 for the next substep.
        const bool last = (t == T - 1) && (k == SIM - 1);
        const float* X1next;
        if (k < SIM - 1) {
          X1next = X1pp[c & 1] + (size_t)tt * B * A1;          // same t
        } else {
          const int tn = t + 1 < T ? t + 1 : 0;                 // next t
          X1next = X1pp[(tn / TCH) & 1] + (size_t)(tn % TCH) * B * A1;
        }
        dim3 grid(32, 1);  // 8 GEMM blocks (N=512) + 24 LIF-only blocks
        snn_layer3_kernel<<<grid, TPB, 0, stream>>>(
            s2, A2, W3bf, A2, b3, vout,
            out + (size_t)t * B * A3, A2, A3,
            /*firstSub=*/(k == 0) ? 1 : 0,
            /*writeOut=*/(k == SIM - 1) ? 1 : 0,
            /*nGemmBlk=*/A3 / 64,
            X1next, v1, s1, B * A1, /*doLif=*/last ? 0 : 1);
      }
    }
  }
}
